// RNN_50173807952120
// MI455X (gfx1250) — matrix operand order repaired and checked by reading, not compiled, not run on hardware
//
#include <hip/hip_runtime.h>
#include <hip/hip_bf16.h>

// ---------------- types ----------------
typedef __attribute__((ext_vector_type(16))) __bf16 v16bf;
typedef __attribute__((ext_vector_type(8)))  float  v8f;
typedef unsigned int        v4ui __attribute__((ext_vector_type(4)));
typedef int                 v8si __attribute__((ext_vector_type(8)));
typedef int                 v4si __attribute__((ext_vector_type(4)));

#define B_  64
#define T_  1024
#define X_  512
#define S_  512
#define U_  64
#define Y_  128
#define KX  1088           // 512 (W) + 512 (Bs) + 64 (Bu)
#define AST 1096           // A_lds row stride (elems), 16B-aligned, conflict-free
#define NRES 4             // K-blocks of BigB resident in LDS

// LDS carve (bytes)
#define RES_BYTES   (NRES * 512 * 80)          // 163840: [kb][n(512), 80B stride][4x16B]
#define ALDS_OFF    RES_BYTES                  // [16][AST] bf16 = 35072
#define STAGE_OFF   (ALDS_OFF + 16 * AST * 2)  // 198912: double-buffer 2x16x512 f32
#define SMEM_BYTES  (STAGE_OFF + 2 * 16 * 512 * 4)  // 264448 < 320KB WGP LDS

// ---------------- helpers ----------------
__device__ __forceinline__ v16bf make_frag(uint4 lo, uint4 hi) {
  union { uint4 q[2]; v16bf v; } u;
  u.q[0] = lo; u.q[1] = hi;
  return u.v;
}

__device__ __forceinline__ uint4 cvt8_bf16(float4 a, float4 b) {
  union { __bf16 h[8]; uint4 u; } r;
  r.h[0] = (__bf16)a.x; r.h[1] = (__bf16)a.y; r.h[2] = (__bf16)a.z; r.h[3] = (__bf16)a.w;
  r.h[4] = (__bf16)b.x; r.h[5] = (__bf16)b.y; r.h[6] = (__bf16)b.z; r.h[7] = (__bf16)b.w;
  return r.u;
}

// Tensor Data Mover: async 2D tile load (16 rows x 512 f32, row stride T_*S_) -> LDS.
// D# built per CDNA5 ISA (group0/group1 bitfields).
// amdgpu-toolchain (clang-23) 6-arg form: (v4u, v8i, v4i, v4i, v8i, i32 cpol).
__device__ __forceinline__ void tdm_load_s(unsigned lds_addr, const float* gsrc) {
  unsigned long long ga = (unsigned long long)(size_t)gsrc;
  v4ui g0;
  g0[0] = 1u;                                   // count=1, user descriptor
  g0[1] = lds_addr;                             // lds_addr (bytes)
  g0[2] = (unsigned)(ga & 0xFFFFFFFFu);         // global_addr[31:0]
  g0[3] = (unsigned)((ga >> 32) & 0x1FFFFFFu)   // global_addr[56:32]
          | (2u << 30);                         // type=2 ("image")
  v8si g1;
  g1[0] = (int)(2u << 16);                      // data_size=2 -> 4 bytes
  g1[1] = (int)(512u << 16);                    // tensor_dim0 = 512 (bits 48..63)
  g1[2] = (int)(16u  << 16);                    // tensor_dim1 = 16 rows (bits 80..95)
  g1[3] = (int)(512u << 16);                    // tile_dim0   = 512 (bits 112..127)
  g1[4] = 16;                                   // tile_dim1   = 16
  g1[5] = (int)(T_ * S_);                       // tensor_dim0_stride = 524288 elems
  g1[6] = 0;
  g1[7] = 0;
  v4si z4 = {0, 0, 0, 0};                       // 2D tensor: groups 2/3 unused
  v8si z8 = {0, 0, 0, 0, 0, 0, 0, 0};
  __builtin_amdgcn_tensor_load_to_lds(g0, g1, z4, z4, z8, 0);
}

// ---------------- kernel 1: pack [W|Bs|Bu] and Wy to bf16 ----------------
__global__ __launch_bounds__(256) void rnn_setup(
    const float* __restrict__ W, const float* __restrict__ Bs,
    const float* __restrict__ Bu, const float* __restrict__ Wy,
    unsigned short* __restrict__ BigB_, unsigned short* __restrict__ WyB_) {
  __bf16* BigB = (__bf16*)BigB_;
  __bf16* WyB  = (__bf16*)WyB_;
  int i = blockIdx.x * 256 + threadIdx.x;
  if (i < X_ * KX) {
    int n = i / KX, k = i % KX;
    float v = (k < 512) ? W[n * 512 + k]
            : (k < 1024) ? Bs[n * 512 + (k - 512)]
                         : Bu[n * 64 + (k - 1024)];
    BigB[i] = (__bf16)v;
  } else {
    int j = i - X_ * KX;
    if (j < Y_ * X_) WyB[j] = (__bf16)Wy[j];
  }
}

// ---------------- kernel 2: sequential scan ----------------
// grid=4 WGs, each owns 16 batch rows; 8 waves, wave w owns output cols [64w,64w+64).
// Per step: acc[16x512] = [x|s_t|u_t](16x1088,bf16) @ BigB^T, then leaky-tanh update.
__global__ __launch_bounds__(256) void rnn_scan(
    const float* __restrict__ x0, const float* __restrict__ s_seq,
    const float* __restrict__ u_seq, const float* __restrict__ bvec,
    const unsigned short* __restrict__ BigB_, float* __restrict__ x_seq) {
  extern __shared__ char smem[];
  const __bf16* BigB = (const __bf16*)BigB_;
  char*   res   = smem;                          // resident BigB K-blocks
  __bf16* A_lds = (__bf16*)(smem + ALDS_OFF);    // [16][AST]
  float*  stage = (float*)(smem + STAGE_OFF);    // [2][16][512] s-tile staging

  const int tid  = threadIdx.x;
  const int lane = tid & 31;
  const int wv   = tid >> 5;        // 0..7
  const int ln   = lane & 15;
  const int kh   = lane >> 4;       // 0/1
  const int b0   = blockIdx.x * 16;

  // Fill resident BigB blocks: res[kb][n] holds 64B (32 K bf16) at 80B stride.
  for (int c = tid; c < NRES * 512 * 4; c += 256) {
    int h = c & 3, n = (c >> 2) & 511, kb = c >> 11;
    uint4 v = *(const uint4*)(BigB + (size_t)n * KX + kb * 32 + h * 8);
    *(uint4*)(res + kb * 40960 + n * 80 + h * 16) = v;
  }

  // x state lives in registers in the C-fragment layout; also emit t=0 output
  float xs[4][8];
  float bias[4];
  for (int tl = 0; tl < 4; ++tl) {
    const int col = wv * 64 + tl * 16 + ln;
    bias[tl] = bvec[col];
    for (int r = 0; r < 8; ++r) {
      const int M = r + 8 * kh;
      float v = x0[(size_t)(b0 + M) * X_ + col];
      xs[tl][r] = v;
      x_seq[((size_t)(b0 + M) * T_ + 0) * X_ + col] = v;
      A_lds[M * AST + col] = (__bf16)v;
    }
  }

  // Kick off TDM load of s tile for t=0 (buffer 0).
  if (tid == 0)
    tdm_load_s((unsigned)(size_t)stage, s_seq + ((size_t)b0 * T_ + 0) * S_);
  __syncthreads();

  for (int t = 0; t < T_ - 1; ++t) {
    __builtin_amdgcn_s_wait_tensorcnt(0);   // s tile for step t landed in LDS
    __syncthreads();
    if (tid == 0 && (t + 1) < T_ - 1)       // prefetch next step into other buffer
      tdm_load_s((unsigned)(size_t)(stage + ((t + 1) & 1) * 16 * 512),
                 s_seq + ((size_t)b0 * T_ + (t + 1)) * S_);

    // stage(f32) -> A_lds cols [512,1024) as bf16
    const float* st = stage + (t & 1) * 16 * 512;
    for (int e = tid; e < 16 * 512; e += 256)
      A_lds[(e >> 9) * AST + 512 + (e & 511)] = (__bf16)st[e];
    // u_t -> A_lds cols [1024,1088)
    for (int e = tid; e < 16 * 64; e += 256) {
      int r = e >> 6, k = e & 63;
      A_lds[r * AST + 1024 + k] =
          (__bf16)u_seq[((size_t)(b0 + r) * T_ + t) * U_ + k];
    }
    __syncthreads();

    v8f acc[4] = {};
    // resident K-blocks (LDS B operands)
    for (int kb = 0; kb < NRES; ++kb) {
      const __bf16* ap = A_lds + ln * AST + kb * 32 + kh * 8;
      v16bf af = make_frag(*(const uint4*)ap, *(const uint4*)(ap + 16));
      for (int tl = 0; tl < 4; ++tl) {
        const int ng = wv * 64 + tl * 16 + ln;
        const char* bp = res + kb * 40960 + ng * 80 + kh * 16;
        v16bf bf = make_frag(*(const uint4*)bp, *(const uint4*)(bp + 32));
        acc[tl] = __builtin_amdgcn_wmma_f32_16x16x32_bf16(
            false, af, false, bf, (short)0, acc[tl], false, false);
      }
    }
    // streamed K-blocks (BigB stays hot in 192MB L2)
    for (int kb = NRES; kb < KX / 32; ++kb) {
      const __bf16* ap = A_lds + ln * AST + kb * 32 + kh * 8;
      v16bf af = make_frag(*(const uint4*)ap, *(const uint4*)(ap + 16));
      for (int tl = 0; tl < 4; ++tl) {
        const int ng = wv * 64 + tl * 16 + ln;
        const __bf16* bp = BigB + (size_t)ng * KX + kb * 32 + kh * 8;
        v16bf bf = make_frag(*(const uint4*)bp, *(const uint4*)(bp + 16));
        acc[tl] = __builtin_amdgcn_wmma_f32_16x16x32_bf16(
            false, af, false, bf, (short)0, acc[tl], false, false);
      }
    }

    // leaky tanh update in fp32, emit x_{t+1}
    for (int tl = 0; tl < 4; ++tl) {
      const int col = wv * 64 + tl * 16 + ln;
      for (int r = 0; r < 8; ++r) {
        const int M = r + 8 * kh;
        float p  = acc[tl][r] + bias[tl];
        float xn = 0.9f * xs[tl][r] + 0.1f * tanhf(p);
        xs[tl][r] = xn;
        x_seq[((size_t)(b0 + M) * T_ + (t + 1)) * X_ + col] = xn;
      }
    }
    __syncthreads();   // everyone finished reading A_lds x-part
    for (int tl = 0; tl < 4; ++tl) {
      const int col = wv * 64 + tl * 16 + ln;
      for (int r = 0; r < 8; ++r)
        A_lds[(r + 8 * kh) * AST + col] = (__bf16)xs[tl][r];
    }
  }
}

// ---------------- kernel 3: y = x_seq @ Wy^T + by ----------------
// grid = (B*T)/16 WGs; 8 waves, each one 16x16 tile (N=128 total).
__global__ __launch_bounds__(256) void rnn_y(
    const float* __restrict__ x_seq, const unsigned short* __restrict__ WyB_,
    const float* __restrict__ by, float* __restrict__ y) {
  const __bf16* WyB = (const __bf16*)WyB_;
  const int tid = threadIdx.x, lane = tid & 31, wv = tid >> 5;
  const int ln = lane & 15, kh = lane >> 4;
  const size_t m0 = (size_t)blockIdx.x * 16;
  const int ng = wv * 16 + ln;

  v8f acc = {};
  for (int kb = 0; kb < X_ / 32; ++kb) {
    const float* ap = x_seq + (m0 + ln) * X_ + kb * 32 + kh * 8;
    float4 a0 = *(const float4*)ap;
    float4 a1 = *(const float4*)(ap + 4);
    float4 a2 = *(const float4*)(ap + 16);
    float4 a3 = *(const float4*)(ap + 20);
    v16bf af = make_frag(cvt8_bf16(a0, a1), cvt8_bf16(a2, a3));
    const __bf16* bp = WyB + (size_t)ng * X_ + kb * 32 + kh * 8;
    v16bf bf = make_frag(*(const uint4*)bp, *(const uint4*)(bp + 16));
    acc = __builtin_amdgcn_wmma_f32_16x16x32_bf16(
        false, af, false, bf, (short)0, acc, false, false);
  }
  const float bv = by[ng];
  for (int r = 0; r < 8; ++r)
    y[(m0 + r + 8 * kh) * Y_ + ng] = acc[r] + bv;
}

// ---------------- host ----------------
extern "C" void kernel_launch(void* const* d_in, const int* in_sizes, int n_in,
                              void* d_out, int out_size, void* d_ws, size_t ws_size,
                              hipStream_t stream) {
  (void)in_sizes; (void)n_in; (void)out_size; (void)ws_size;
  const float* x0    = (const float*)d_in[0];
  const float* s_seq = (const float*)d_in[1];
  const float* u_seq = (const float*)d_in[2];
  const float* W     = (const float*)d_in[3];
  const float* bvec  = (const float*)d_in[4];
  const float* Bs    = (const float*)d_in[5];
  const float* Bu    = (const float*)d_in[6];
  const float* Wy    = (const float*)d_in[7];
  const float* by    = (const float*)d_in[8];

  float* x_seq = (float*)d_out;                              // [B,T,X]
  float* y_seq = x_seq + (size_t)B_ * T_ * X_;               // [B,T,Y]

  unsigned short* BigB = (unsigned short*)d_ws;              // 512*1088 bf16
  unsigned short* WyB  = (unsigned short*)((char*)d_ws + (size_t)X_ * KX * 2);

  // 1) pack weights to bf16 (BigB = [W|Bs|Bu], WyB)
  {
    int total = X_ * KX + Y_ * X_;
    rnn_setup<<<(total + 255) / 256, 256, 0, stream>>>(W, Bs, Bu, Wy, BigB, WyB);
  }
  // 2) sequential scan: 4 WGs x 16 batch rows, no inter-WG sync needed
  (void)hipFuncSetAttribute((const void*)rnn_scan,
                            hipFuncAttributeMaxDynamicSharedMemorySize,
                            SMEM_BYTES);
  rnn_scan<<<4, 256, SMEM_BYTES, stream>>>(x0, s_seq, u_seq, bvec, BigB, x_seq);
  // 3) output projection
  rnn_y<<<(B_ * T_) / 16, 256, 0, stream>>>(x_seq, WyB, by, y_seq);
}
